// HeteroGraphSAGE_45792941310085
// MI455X (gfx1250) — matrix operand order repaired and checked by reading, not compiled, not run on hardware
//
#include <hip/hip_runtime.h>

typedef __attribute__((ext_vector_type(16))) __bf16 v16bf;
typedef __attribute__((ext_vector_type(8)))  float  v8f;

#define NN 100000
#define EE 300000
#define CC 256
#define NCF (NN * CC)

// ---------------------------------------------------------------- zero scratch
__global__ void zero_f32(float* __restrict__ p, int n) {
  int i = blockIdx.x * blockDim.x + threadIdx.x;
  if (i < n) p[i] = 0.0f;
}

// ------------------------------------------------- pack weights to B-fragments
// Combined B[k][j], k in [0,512): k<256 -> Wl[l][j][k] ; k>=256 -> Wr[l][j][k-256]
// Fragment order: pw[dl*131072 + ((w*16+ks)*32 + lane)*16 + i]
//   j = w*16 + (lane&15), k = ks*32 + (lane>>4)*16 + i  (contiguous k order, same as the A frag)
__global__ void pack_w(const float* __restrict__ Wl_u2i, const float* __restrict__ Wr_u2i,
                       const float* __restrict__ Wl_i2u, const float* __restrict__ Wr_i2u,
                       __bf16* __restrict__ pw) {
  int e = blockIdx.x * blockDim.x + threadIdx.x;   // < 4*131072
  int dl   = e >> 17;
  int rem  = e & 0x1FFFF;
  int i    = rem & 15;
  int lane = (rem >> 4) & 31;
  int ks   = (rem >> 9) & 15;
  int w    = rem >> 13;
  int j = w * 16 + (lane & 15);
  int k = ks * 32 + (lane >> 4) * 16 + i;
  int l = dl >> 1;
  const float* Wl = (dl & 1) ? Wl_i2u : Wl_u2i;
  const float* Wr = (dl & 1) ? Wr_i2u : Wr_u2i;
  float v = (k < CC) ? Wl[l * CC * CC + j * CC + k]
                     : Wr[l * CC * CC + j * CC + (k - CC)];
  pw[e] = (__bf16)v;
}

// ------------------------------------------------------- edge scatter (seg sum)
__global__ void scatter_accum(const float* __restrict__ xsrc, const int* __restrict__ ei,
                              float* __restrict__ agg, float* __restrict__ cnt) {
  int e   = blockIdx.x;            // one block per edge
  int src = ei[e];
  int dst = ei[EE + e];
  int t   = threadIdx.x;           // 256 threads = one channel each
  atomicAdd(&agg[(size_t)dst * CC + t], xsrc[(size_t)src * CC + t]);
  if (t == 0) atomicAdd(&cnt[dst], 1.0f);
}

// ---------------------------- fused mean-build + dual GEMM + bias + LN + ReLU
__global__ __launch_bounds__(512)
void sage_gemm_ln(const float* __restrict__ agg, const float* __restrict__ cnt,
                  const float* __restrict__ xdst, const __bf16* __restrict__ pw,
                  const float* __restrict__ bias, const float* __restrict__ gamma,
                  const float* __restrict__ beta, float* __restrict__ out) {
  __shared__ __align__(16) __bf16 As[16][520];   // 16 rows x 512 K (+8 pad)
  __shared__ float Os[16][260];                  // 16 rows x 256 out (+4 pad)
  __shared__ float rinv[16];

  const int t    = threadIdx.x;
  const int row0 = blockIdx.x * 16;

  if (t < 16) rinv[t] = 1.0f / fmaxf(cnt[row0 + t], 1.0f);
  __syncthreads();

  // A tile: K 0..255 = mean = agg*rinv ; K 256..511 = x_dst   (f32 -> bf16)
  for (int idx = t; idx < 16 * 512; idx += 512) {
    int r   = idx >> 9;
    int col = idx & 511;
    size_t gr = (size_t)(row0 + r);
    float v = (col < CC) ? agg[gr * CC + col] * rinv[r]
                         : xdst[gr * CC + (col - CC)];
    As[r][col] = (__bf16)v;
  }
  __syncthreads();

  const int w     = t >> 5;        // wave -> output col tile [16w, 16w+16)
  const int lane  = t & 31;
  const int r     = lane & 15;
  const int hi    = lane >> 4;
  const int khalf = hi * 16;

  v8f acc = {};
  const __bf16* pwW = pw + (size_t)w * 8192;
  #pragma unroll
  for (int ks = 0; ks < 16; ++ks) {
    v16bf a, b;
    // A frag: lane holds row r, K = khalf..khalf+15 (contiguous, matches packed B)
    ((uint4*)&a)[0] = *(const uint4*)&As[r][ks * 32 + khalf];
    ((uint4*)&a)[1] = *(const uint4*)&As[r][ks * 32 + khalf + 8];
    // B frag: pre-packed, 32 contiguous bytes per lane
    const __bf16* bp = pwW + (size_t)(ks * 32 + lane) * 16;
    ((uint4*)&b)[0] = ((const uint4*)bp)[0];
    ((uint4*)&b)[1] = ((const uint4*)bp)[1];
    acc = __builtin_amdgcn_wmma_f32_16x16x32_bf16(false, a, false, b,
                                                  (short)0, acc, false, false);
  }

  // C/D layout: VGPR v, lanes 0-15 -> M=v, lanes 16-31 -> M=v+8 ; N = lane&15
  const float bj = bias[w * 16 + r];
  #pragma unroll
  for (int v = 0; v < 8; ++v)
    Os[v + 8 * hi][w * 16 + r] = acc[v] + bj;
  __syncthreads();

  // LayerNorm + ReLU: wave w normalizes row w (wave32 reduction)
  float s = 0.f, ss = 0.f;
  #pragma unroll
  for (int c = lane; c < CC; c += 32) {
    float x = Os[w][c];
    s += x; ss += x * x;
  }
  #pragma unroll
  for (int off = 16; off >= 1; off >>= 1) {
    s  += __shfl_xor(s, off, 32);
    ss += __shfl_xor(ss, off, 32);
  }
  float mu  = s * (1.0f / CC);
  float var = fmaxf(ss * (1.0f / CC) - mu * mu, 0.0f);
  float rs  = rsqrtf(var + 1e-5f);
  size_t gr = (size_t)(row0 + w);
  #pragma unroll
  for (int c = lane; c < CC; c += 32) {
    float y = (Os[w][c] - mu) * rs * gamma[c] + beta[c];
    out[gr * CC + c] = fmaxf(y, 0.0f);
  }
}

// ----------------------------------------------------------------------- host
extern "C" void kernel_launch(void* const* d_in, const int* in_sizes, int n_in,
                              void* d_out, int out_size, void* d_ws, size_t ws_size,
                              hipStream_t stream) {
  (void)in_sizes; (void)n_in; (void)out_size; (void)ws_size;

  const float* x_user = (const float*)d_in[0];
  const float* x_item = (const float*)d_in[1];
  const int*   ei_u2i = (const int*)d_in[2];
  const int*   ei_i2u = (const int*)d_in[3];
  const float* Wl_u2i = (const float*)d_in[4];
  const float* bl_u2i = (const float*)d_in[5];
  const float* Wr_u2i = (const float*)d_in[6];
  const float* Wl_i2u = (const float*)d_in[7];
  const float* bl_i2u = (const float*)d_in[8];
  const float* Wr_i2u = (const float*)d_in[9];
  const float* g_user = (const float*)d_in[10];
  const float* b_user = (const float*)d_in[11];
  const float* g_item = (const float*)d_in[12];
  const float* b_item = (const float*)d_in[13];

  float* out_user = (float*)d_out;                       // stack([user, item])
  float* out_item = (float*)d_out + (size_t)NCF;

  char* ws = (char*)d_ws;
  size_t off = 0;
  __bf16* pw = (__bf16*)(ws + off); off += (size_t)4 * 131072 * sizeof(__bf16);
  off = (off + 255) & ~(size_t)255;
  float* agg = (float*)(ws + off);  off += (size_t)NCF * 4;
  float* cnt = (float*)(ws + off);  off += (size_t)NN * 4;   // contiguous after agg
  off = (off + 255) & ~(size_t)255;
  float* u1  = (float*)(ws + off);  off += (size_t)NCF * 4;
  float* i1  = (float*)(ws + off);  off += (size_t)NCF * 4;

  pack_w<<<(4 * 131072) / 256, 256, 0, stream>>>(Wl_u2i, Wr_u2i, Wl_i2u, Wr_i2u, pw);

  auto run = [&](const float* xsrc, const float* xd, const int* ei, int dl,
                 const float* bl, int l, const float* g, const float* b, float* o) {
    int nz = NCF + NN;                                   // agg + cnt in one pass
    zero_f32<<<(nz + 255) / 256, 256, 0, stream>>>(agg, nz);
    scatter_accum<<<EE, 256, 0, stream>>>(xsrc, ei, agg, cnt);
    sage_gemm_ln<<<NN / 16, 512, 0, stream>>>(agg, cnt, xd, pw + (size_t)dl * 131072,
                                              bl + l * CC, g + l * CC, b + l * CC, o);
  };

  // layer 0
  run(x_user, x_item, ei_u2i, 0, bl_u2i, 0, g_item, b_item, i1);  // new item
  run(x_item, x_user, ei_i2u, 1, bl_i2u, 0, g_user, b_user, u1);  // new user
  // layer 1 -> final output
  run(u1, i1, ei_u2i, 2, bl_u2i, 1, g_item, b_item, out_item);
  run(i1, u1, ei_i2u, 3, bl_i2u, 1, g_user, b_user, out_user);
}
